// PartialRoPEAttention_21921513079365
// MI455X (gfx1250) — compile-verified, matrix operand order checked by reading, not run on hardware
//
#include <hip/hip_runtime.h>
#include <hip/hip_bf16.h>

#define DIM      2048
#define SEQ      2048
#define N_HEADS  16
#define N_KVH    4
#define HEAD_DIM 128
#define KV_DIM   512
#define ROPE_D   16

typedef unsigned short u16;
typedef __attribute__((ext_vector_type(16))) __bf16 v16bf;
typedef __attribute__((ext_vector_type(8)))  float  v8f;

// ---------- CDNA5 async memory->LDS path (guarded; sync copy fallback) ----------
#if defined(__has_builtin)
#if __has_builtin(__builtin_amdgcn_global_load_async_to_lds_b128) && \
    __has_builtin(__builtin_amdgcn_s_wait_asynccnt)
#define ASYNC_COPY 1
#endif
#if __has_builtin(__builtin_amdgcn_tensor_load_to_lds) && \
    __has_builtin(__builtin_amdgcn_s_wait_tensorcnt)
#define TDM_COPY 1
#endif
#endif
#ifndef ASYNC_COPY
#define ASYNC_COPY 0
#endif
#ifndef TDM_COPY
#define TDM_COPY 0
#endif

#if ASYNC_COPY
// Builtin parameter type (from compiler diagnostic): vector_size(16) int*,
// global (__device__) source / LDS (__shared__) destination; generic pointers
// with the exact pointee type convert under HIP rules.
typedef int v4i_ __attribute__((vector_size(4 * sizeof(int))));
#endif

#if TDM_COPY
typedef unsigned int u32x4_ __attribute__((ext_vector_type(4)));
typedef int          i32x8_ __attribute__((ext_vector_type(8)));
typedef int          i32x4_ __attribute__((ext_vector_type(4)));

// One TDM descriptor: load a (rows x 32) bf16 tile, row stride
// `row_stride_elems`, packed contiguously into LDS at `lds_dst`.
// D# layout per CDNA5 ISA 8.3/8.4 (group0: count/lds/global/type,
// group1: data_size, tensor dims, tile dims, strides). Groups 2/3 zero (2-D).
// This toolchain's builtin is the 6-arg form:
//   (u32x4 g0, i32x8 g1, i32x4 g2, i32x4 g3, i32x8 extra, i32 cpol)
__device__ __forceinline__ void tdm_load_2d(const u16* gsrc, u16* lds_dst,
                                            unsigned int rows,
                                            unsigned int row_stride_elems) {
    unsigned long long ga = (unsigned long long)gsrc;
    unsigned int lo = (unsigned int)(unsigned long long)lds_dst; // low 32b = LDS byte offset

    u32x4_ g0;
    g0[0] = 1u;                                            // count=1 (valid user D#)
    g0[1] = lo;                                            // lds_addr
    g0[2] = (unsigned int)ga;                              // global_addr[31:0]
    g0[3] = (unsigned int)((ga >> 32) & 0x1FFFFFFu)        // global_addr[56:32]
          | (2u << 30);                                    // type = 2 ("image")

    const unsigned int td0 = 32u;                          // tensor_dim0 (elems)
    const unsigned int td1 = rows;                         // tensor_dim1
    const unsigned long long st0 = row_stride_elems;       // tensor_dim0_stride

    i32x8_ g1;
    g1[0] = (int)(1u << 16);                               // data_size=1 (2 bytes)
    g1[1] = (int)((td0 & 0xFFFFu) << 16);                  // tensor_dim0[15:0] @ bit48
    g1[2] = (int)((td0 >> 16) | ((td1 & 0xFFFFu) << 16));  // dim0[31:16] | dim1[15:0]
    g1[3] = (int)((td1 >> 16) | (32u << 16));              // dim1[31:16] | tile_dim0=32
    g1[4] = (int)(td1 & 0xFFFFu);                          // tile_dim1=rows, tile_dim2=0
    g1[5] = (int)(st0 & 0xFFFFFFFFu);                      // dim0_stride[31:0]
    g1[6] = (int)((st0 >> 32) & 0xFFFFu);                  // dim0_stride[47:32] | dim1_stride[15:0]=0
    g1[7] = 0;                                             // dim1_stride[47:16]=0

    i32x4_ z4 = {0, 0, 0, 0};
    i32x8_ z8 = {0, 0, 0, 0, 0, 0, 0, 0};
    __builtin_amdgcn_tensor_load_to_lds(g0, g1, z4, z4, z8, 0);
}
#endif

// ---------- helpers ----------

__device__ __forceinline__ u16 f2bf(float f) {
    unsigned int u = __float_as_uint(f);
    u += 0x7FFFu + ((u >> 16) & 1u);   // round-to-nearest-even
    return (u16)(u >> 16);
}

__device__ __forceinline__ v8f wmma_bf16(v16bf a, v16bf b, v8f c) {
    return __builtin_amdgcn_wmma_f32_16x16x32_bf16(
        /*neg_a=*/false, a, /*neg_b=*/false, b,
        /*c_mod=*/(short)0, c, /*reuse_a=*/false, /*reuse_b=*/false);
}

// Load a 16x32 bf16 fragment (A or B^T) from row-major memory, leading dim `ld`
// (elements). ISA 7.12.2 layout for 16-bit A:
//   lanes 0-15 : row m=lane,    K = {0..7, 16..23}
//   lanes 16-31: row m=lane-16, K = {8..15, 24..31}
__device__ __forceinline__ v16bf load_frag(const u16* p, int ld) {
    const int lane = threadIdx.x & 31;
    const int m  = lane & 15;
    const int kb = (lane & 16) >> 1;   // 0 or 8
    union { v16bf v; uint4 q[2]; } u;
    u.q[0] = *reinterpret_cast<const uint4*>(p + m * ld + kb);
    u.q[1] = *reinterpret_cast<const uint4*>(p + m * ld + kb + 16);
    return u.v;
}

// ---------- kernels ----------

__global__ __launch_bounds__(256) void k_f32_to_bf16(const float* __restrict__ src,
                                                     u16* __restrict__ dst, int n) {
    int i = blockIdx.x * blockDim.x + threadIdx.x;
    if (i < n) dst[i] = f2bf(src[i]);
}

// C[M,N](f32) = A[M,K](bf16) * B[N,K](bf16)^T
// block = 128 threads (4 waves), block tile 128(M) x 128(N), K step 32,
// double-buffered LDS staged by the Tensor Data Mover (TDM) when available
// (wave0 -> A tile, wave1 -> B tile), else per-lane async loads, else sync.
// Waves in a 2x2 grid; each wave computes a 64x64 tile = 4x4 WMMA tiles.
__global__ __launch_bounds__(128) void k_gemm_bf16(const u16* __restrict__ A,
                                                   const u16* __restrict__ B,
                                                   float* __restrict__ C,
                                                   int M, int N, int K) {
    __shared__ __align__(16) u16 As[2][128 * 32];
    __shared__ __align__(16) u16 Bs[2][128 * 32];

    const int tid  = threadIdx.x;
    const int lane = tid & 31;
    const int wid  = tid >> 5;
    const int wy   = wid >> 1;          // 0..1
    const int wx   = wid & 1;           // 0..1
    const int m0   = blockIdx.y * 128;
    const int n0   = blockIdx.x * 128;

    // stage one 128x32 tile of A and of B into LDS buffer `buf`
    auto stage = [&](int buf, int k0) {
#if TDM_COPY
        if (wid == 0)
            tdm_load_2d(A + (size_t)m0 * K + k0, &As[buf][0], 128u, (unsigned)K);
        else if (wid == 1)
            tdm_load_2d(B + (size_t)n0 * K + k0, &Bs[buf][0], 128u, (unsigned)K);
#else
        #pragma unroll
        for (int i = 0; i < 4; ++i) {
            int c   = tid + 128 * i;          // 0..511
            int row = c >> 2, col = (c & 3) << 3;
#if ASYNC_COPY
            __builtin_amdgcn_global_load_async_to_lds_b128(
                (v4i_*)(A + (size_t)(m0 + row) * K + k0 + col),
                (v4i_*)(&As[buf][row * 32 + col]), 0, 0);
            __builtin_amdgcn_global_load_async_to_lds_b128(
                (v4i_*)(B + (size_t)(n0 + row) * K + k0 + col),
                (v4i_*)(&Bs[buf][row * 32 + col]), 0, 0);
#else
            *reinterpret_cast<uint4*>(&As[buf][row * 32 + col]) =
                *reinterpret_cast<const uint4*>(&A[(size_t)(m0 + row) * K + k0 + col]);
            *reinterpret_cast<uint4*>(&Bs[buf][row * 32 + col]) =
                *reinterpret_cast<const uint4*>(&B[(size_t)(n0 + row) * K + k0 + col]);
#endif
        }
#endif
    };

    v8f acc[4][4] = {};

    stage(0, 0);
    for (int k0 = 0; k0 < K; k0 += 32) {
        const int  cur      = (k0 >> 5) & 1;
        const bool has_next = (k0 + 32) < K;
        if (has_next) stage(cur ^ 1, k0 + 32);   // prefetch next buffer
#if TDM_COPY
        if (wid < 2) {
            if (has_next) __builtin_amdgcn_s_wait_tensorcnt(1); // keep prefetch in flight
            else          __builtin_amdgcn_s_wait_tensorcnt(0);
        }
#elif ASYNC_COPY
        if (has_next) __builtin_amdgcn_s_wait_asynccnt(8);      // keep next 8 in flight
        else          __builtin_amdgcn_s_wait_asynccnt(0);
#endif
        __syncthreads();

        v16bf af[4], bfv[4];
        #pragma unroll
        for (int i = 0; i < 4; ++i)
            af[i]  = load_frag(&As[cur][(wy * 64 + i * 16) * 32], 32);
        #pragma unroll
        for (int j = 0; j < 4; ++j)
            bfv[j] = load_frag(&Bs[cur][(wx * 64 + j * 16) * 32], 32);

        #pragma unroll
        for (int i = 0; i < 4; ++i)
            #pragma unroll
            for (int j = 0; j < 4; ++j)
                acc[i][j] = wmma_bf16(af[i], bfv[j], acc[i][j]);

        __syncthreads();
    }

    const int hlf = lane >> 4;
    #pragma unroll
    for (int i = 0; i < 4; ++i)
        #pragma unroll
        for (int j = 0; j < 4; ++j)
            #pragma unroll
            for (int r = 0; r < 8; ++r) {
                int row = m0 + wy * 64 + i * 16 + r + 8 * hlf;
                int col = n0 + wx * 64 + j * 16 + (lane & 15);
                C[(size_t)row * N + col] = acc[i][j][r];
            }
}

// Per-row epilogue: RMS-norm each head vector, partial RoPE (dims 0..15),
// q_gain, then pack Q/K/V to bf16. grid = SEQ, block = 256 (8 waves).
// Lane L of a wave holds dims {L, L+32, L+64, L+96} of one head vector.
__global__ __launch_bounds__(256) void k_qkv_epilogue(const float* __restrict__ Qf,
                                                      const float* __restrict__ Kf,
                                                      const float* __restrict__ Vf,
                                                      const float* __restrict__ gain,
                                                      u16* __restrict__ Qb,
                                                      u16* __restrict__ Kb,
                                                      u16* __restrict__ Vb) {
    const int s    = blockIdx.x;
    const int tid  = threadIdx.x;
    const int lane = tid & 31;
    const int wave = tid >> 5;

    const int   fi  = lane & 7;
    const float ang = (float)s * powf(10000.0f, -(float)fi * 0.125f);
    const float ca  = cosf(ang), sa = sinf(ang);

    for (int h = wave; h < N_HEADS; h += 8) {
        const float* base = Qf + (size_t)s * DIM + h * HEAD_DIM;
        float v[4];
        float ss = 0.f;
        #pragma unroll
        for (int i = 0; i < 4; ++i) { v[i] = base[lane + 32 * i]; ss += v[i] * v[i]; }
        #pragma unroll
        for (int d = 1; d < 32; d <<= 1) ss += __shfl_xor(ss, d, 32);
        float rn = rsqrtf(ss * (1.0f / HEAD_DIM) + 1.1920929e-7f);
        #pragma unroll
        for (int i = 0; i < 4; ++i) v[i] *= rn;
        float part = __shfl_xor(v[0], 8, 32);
        if (lane < ROPE_D)
            v[0] = (lane < 8) ? (v[0] * ca + part * sa) : (-part * sa + v[0] * ca);
        float g = gain[h];
        u16* ob = Qb + (size_t)s * DIM + h * HEAD_DIM;
        #pragma unroll
        for (int i = 0; i < 4; ++i) ob[lane + 32 * i] = f2bf(v[i] * g);
    }

    for (int h = wave; h < N_KVH; h += 8) {
        const float* base = Kf + (size_t)s * KV_DIM + h * HEAD_DIM;
        float v[4];
        float ss = 0.f;
        #pragma unroll
        for (int i = 0; i < 4; ++i) { v[i] = base[lane + 32 * i]; ss += v[i] * v[i]; }
        #pragma unroll
        for (int d = 1; d < 32; d <<= 1) ss += __shfl_xor(ss, d, 32);
        float rn = rsqrtf(ss * (1.0f / HEAD_DIM) + 1.1920929e-7f);
        #pragma unroll
        for (int i = 0; i < 4; ++i) v[i] *= rn;
        float part = __shfl_xor(v[0], 8, 32);
        if (lane < ROPE_D)
            v[0] = (lane < 8) ? (v[0] * ca + part * sa) : (-part * sa + v[0] * ca);
        u16* ob = Kb + (size_t)s * KV_DIM + h * HEAD_DIM;
        #pragma unroll
        for (int i = 0; i < 4; ++i) ob[lane + 32 * i] = f2bf(v[i]);
    }

    for (int idx = tid; idx < KV_DIM; idx += 256)
        Vb[(size_t)s * KV_DIM + idx] = f2bf(Vf[(size_t)s * KV_DIM + idx]);
}

// Flash attention (causal, GQA 4:1). grid = (SEQ/64, N_HEADS), block = 128 (4 waves).
// Each wave owns a 16-query tile; K/V staged 32 keys at a time in LDS.
__global__ __launch_bounds__(128) void k_flash_attn(const u16* __restrict__ Qb,
                                                    const u16* __restrict__ Kb,
                                                    const u16* __restrict__ Vb,
                                                    u16* __restrict__ Yb) {
    __shared__ __align__(16) u16 Ks[32 * 128];      // [key][d]
    __shared__ __align__(16) u16 Vt[128 * 32];      // [d][key] (transposed)
    __shared__ __align__(16) u16 Ps[4][16 * 32];    // per-wave P tile [q][key]

    const int tid  = threadIdx.x;
    const int lane = tid & 31;
    const int wave = tid >> 5;
    const int h    = blockIdx.y;
    const int kvh  = h >> 2;
    const int qblk = blockIdx.x * 64;
    const int q0   = qblk + wave * 16;
    const int hlf  = lane >> 4;
    const int keyn = lane & 15;
    const float SCALE = 0.08838834764831845f;   // 1/sqrt(128)

    // Q fragments, 4 K-chunks of 32 (A layout)
    v16bf aq[4];
    {
        const int m  = lane & 15;
        const int kb = (lane & 16) >> 1;
        const u16* qp = Qb + (size_t)(q0 + m) * DIM + h * HEAD_DIM;
        #pragma unroll
        for (int c4 = 0; c4 < 4; ++c4) {
            union { v16bf v; uint4 q[2]; } u;
            u.q[0] = *reinterpret_cast<const uint4*>(qp + c4 * 32 + kb);
            u.q[1] = *reinterpret_cast<const uint4*>(qp + c4 * 32 + kb + 16);
            aq[c4] = u.v;
        }
    }

    v8f o[8] = {};
    float mi[8], li[8];
    #pragma unroll
    for (int r = 0; r < 8; ++r) { mi[r] = -1e30f; li[r] = 0.f; }

    for (int j0 = 0; j0 <= qblk + 63; j0 += 32) {
        // K tile: 32x128 = 512 x 16B, 4 per thread
        #pragma unroll
        for (int i = 0; i < 4; ++i) {
            int cidx = tid + 128 * i;
            int row = cidx >> 4, col = (cidx & 15) << 3;
#if ASYNC_COPY
            __builtin_amdgcn_global_load_async_to_lds_b128(
                (v4i_*)(Kb + (size_t)(j0 + row) * KV_DIM + kvh * HEAD_DIM + col),
                (v4i_*)(&Ks[row * 128 + col]), 0, 0);
#else
            *reinterpret_cast<uint4*>(&Ks[row * 128 + col]) =
                *reinterpret_cast<const uint4*>(
                    &Kb[(size_t)(j0 + row) * KV_DIM + kvh * HEAD_DIM + col]);
#endif
        }
        // V tile transposed: coalesced read, scattered LDS write
        #pragma unroll
        for (int i = 0; i < 32; ++i) {
            int e = tid + 128 * i;
            int j = e >> 7, d = e & 127;
            Vt[d * 32 + j] = Vb[(size_t)(j0 + j) * KV_DIM + kvh * HEAD_DIM + d];
        }
#if ASYNC_COPY
        __builtin_amdgcn_s_wait_asynccnt(0);
#endif
        __syncthreads();

        if (j0 <= q0 + 15) {
            // S = Q * K^T : 16q x 32k as two 16x16 tiles, K-depth 128
            v8f s0 = {}, s1 = {};
            #pragma unroll
            for (int c4 = 0; c4 < 4; ++c4) {
                v16bf bk0 = load_frag(&Ks[ 0 * 128 + c4 * 32], 128);
                v16bf bk1 = load_frag(&Ks[16 * 128 + c4 * 32], 128);
                s0 = wmma_bf16(aq[c4], bk0, s0);
                s1 = wmma_bf16(aq[c4], bk1, s1);
            }
            // online softmax (rows m = r + 8*hlf, cols = keyn / keyn+16)
            #pragma unroll
            for (int r = 0; r < 8; ++r) {
                int qr = q0 + 8 * hlf + r;
                float v0 = s0[r] * SCALE; if (j0 + keyn      > qr) v0 = -1e30f;
                float v1 = s1[r] * SCALE; if (j0 + 16 + keyn > qr) v1 = -1e30f;
                float mx = fmaxf(v0, v1);
                #pragma unroll
                for (int d = 1; d < 16; d <<= 1) mx = fmaxf(mx, __shfl_xor(mx, d, 32));
                float mnew  = fmaxf(mi[r], mx);
                float alpha = __expf(mi[r] - mnew);
                float p0 = __expf(v0 - mnew), p1 = __expf(v1 - mnew);
                float rs = p0 + p1;
                #pragma unroll
                for (int d = 1; d < 16; d <<= 1) rs += __shfl_xor(rs, d, 32);
                li[r] = li[r] * alpha + rs;
                mi[r] = mnew;
                #pragma unroll
                for (int t = 0; t < 8; ++t) o[t][r] = o[t][r] * alpha;
                int mrow = r + 8 * hlf;
                Ps[wave][mrow * 32 + keyn]      = f2bf(p0);
                Ps[wave][mrow * 32 + 16 + keyn] = f2bf(p1);
            }
            // O += P * V (one K=32 WMMA per 16-wide d tile)
            v16bf pf = load_frag(&Ps[wave][0], 32);
            #pragma unroll
            for (int t = 0; t < 8; ++t) {
                v16bf bv = load_frag(&Vt[(t * 16) * 32], 32);
                o[t] = wmma_bf16(pf, bv, o[t]);
            }
        }
        __syncthreads();
    }

    #pragma unroll
    for (int t = 0; t < 8; ++t)
        #pragma unroll
        for (int r = 0; r < 8; ++r) {
            int srow = q0 + r + 8 * hlf;
            int d    = t * 16 + keyn;
            Yb[(size_t)srow * DIM + h * HEAD_DIM + d] = f2bf(o[t][r] / li[r]);
        }
}

// ---------- launch ----------

extern "C" void kernel_launch(void* const* d_in, const int* in_sizes, int n_in,
                              void* d_out, int out_size, void* d_ws, size_t ws_size,
                              hipStream_t stream) {
    (void)in_sizes; (void)n_in; (void)out_size; (void)ws_size;

    const float* x  = (const float*)d_in[0];
    const float* Wq = (const float*)d_in[1];
    const float* Wk = (const float*)d_in[2];
    const float* Wv = (const float*)d_in[3];
    const float* Wo = (const float*)d_in[4];
    const float* qg = (const float*)d_in[5];
    float* out = (float*)d_out;

    size_t off = 0;
    auto wsa = [&](size_t bytes) {
        char* p = (char*)d_ws + off;
        off += (bytes + 255) & ~(size_t)255;
        return (void*)p;
    };
    u16*   xb  = (u16*)  wsa((size_t)SEQ * DIM * 2);
    u16*   Wqb = (u16*)  wsa((size_t)DIM * DIM * 2);
    u16*   Wkb = (u16*)  wsa((size_t)KV_DIM * DIM * 2);
    u16*   Wvb = (u16*)  wsa((size_t)KV_DIM * DIM * 2);
    u16*   Wob = (u16*)  wsa((size_t)DIM * DIM * 2);
    float* Qf  = (float*)wsa((size_t)SEQ * DIM * 4);
    float* Kf  = (float*)wsa((size_t)SEQ * KV_DIM * 4);
    float* Vf  = (float*)wsa((size_t)SEQ * KV_DIM * 4);
    u16*   Qb  = (u16*)  wsa((size_t)SEQ * DIM * 2);
    u16*   Kb  = (u16*)  wsa((size_t)SEQ * KV_DIM * 2);
    u16*   Vb  = (u16*)  wsa((size_t)SEQ * KV_DIM * 2);
    u16*   Yb  = (u16*)  wsa((size_t)SEQ * DIM * 2);

    const int T = 256;
    auto cvt = [&](const float* s, u16* d, int n) {
        k_f32_to_bf16<<<(n + T - 1) / T, T, 0, stream>>>(s, d, n);
    };
    cvt(x,  xb,  SEQ * DIM);
    cvt(Wq, Wqb, DIM * DIM);
    cvt(Wk, Wkb, KV_DIM * DIM);
    cvt(Wv, Wvb, KV_DIM * DIM);
    cvt(Wo, Wob, DIM * DIM);

    // projections
    k_gemm_bf16<<<dim3(DIM / 128,    SEQ / 128), 128, 0, stream>>>(xb, Wqb, Qf, SEQ, DIM,    DIM);
    k_gemm_bf16<<<dim3(KV_DIM / 128, SEQ / 128), 128, 0, stream>>>(xb, Wkb, Kf, SEQ, KV_DIM, DIM);
    k_gemm_bf16<<<dim3(KV_DIM / 128, SEQ / 128), 128, 0, stream>>>(xb, Wvb, Vf, SEQ, KV_DIM, DIM);

    // RMS-norm + partial RoPE + gain + bf16 pack
    k_qkv_epilogue<<<SEQ, 256, 0, stream>>>(Qf, Kf, Vf, qg, Qb, Kb, Vb);

    // causal GQA flash attention
    k_flash_attn<<<dim3(SEQ / 64, N_HEADS), 128, 0, stream>>>(Qb, Kb, Vb, Yb);

    // output projection
    k_gemm_bf16<<<dim3(DIM / 128, SEQ / 128), 128, 0, stream>>>(Yb, Wob, out, SEQ, DIM, DIM);
}
